// SimpleNet_35519379538521
// MI455X (gfx1250) — matrix operand order repaired and checked by reading, not compiled, not run on hardware
//
#include <hip/hip_runtime.h>
#include <hip/hip_bf16.h>

// ---------------------------------------------------------------------------
// Bipartite GNN forward (MI455X / gfx1250, wave32, bf16 WMMA w/ f32 accum).
//
// All GEMMs: M tiled by 16 (one wave per tile), N = 64 (4 WMMA n-tiles),
// K looped in steps of 32 with v_wmma_f32_16x16x32_bf16.
// Activations fit in 192MB L2 -> compute/latency bound, so bf16 matrix pipes.
// ---------------------------------------------------------------------------

#define kNV 100000
#define kNC 50000
#define kE  200000
#define kH  64
#define kEPS 1e-5f

typedef __attribute__((ext_vector_type(16))) __bf16 v16bf;
typedef __attribute__((ext_vector_type(8)))  __bf16 v8bf;
typedef __attribute__((ext_vector_type(8)))  float  v8f;

// fp32 -> bf16, round-to-nearest-even
__device__ __forceinline__ __bf16 f2bf(float f) {
  unsigned u = __float_as_uint(f);
  unsigned short h = (unsigned short)((u + 0x7FFFu + ((u >> 16) & 1u)) >> 16);
  return __builtin_bit_cast(__bf16, h);
}

// A fragment (16-bit A 16x32 layout): lane m = l&15; lanes 0-15 hold K 0-7 /
// 16-23, lanes 16-31 hold K 8-15 / 24-31. rowk points at A row, K offset 0.
__device__ __forceinline__ v16bf load_fragA(const __bf16* rowk, int lane) {
  const int khalf = (lane >> 4) << 3;
  const v8bf lo = *reinterpret_cast<const v8bf*>(rowk + khalf);
  const v8bf hi = *reinterpret_cast<const v8bf*>(rowk + khalf + 16);
  v16bf r;
#pragma unroll
  for (int j = 0; j < 8; ++j) { r[j] = lo[j]; r[8 + j] = hi[j]; }
  return r;
}

// B fragment (16-bit B 32x16 layout): lane n = l&15; lanes 0-15 hold K 0-7 /
// 16-23, lanes 16-31 hold K 8-15 / 24-31. rowk points at transposed-weight row (n-major).
__device__ __forceinline__ v16bf load_fragB(const __bf16* rowk, int lane) {
  const int kb = (lane >> 4) << 3;
  const v8bf lo = *reinterpret_cast<const v8bf*>(rowk + kb);
  const v8bf hi = *reinterpret_cast<const v8bf*>(rowk + kb + 16);
  v16bf r;
#pragma unroll
  for (int j = 0; j < 8; ++j) { r[j] = lo[j]; r[8 + j] = hi[j]; }
  return r;
}

// One 16-row tile x 64-col GEMM: C[16,64] += A[16,KDIM] * Wt[64,KDIM]^T
template <int KDIM>
__device__ __forceinline__ void wmma_tile_gemm(const __bf16* ldsA, int ldA,
                                               const __bf16* ldsWt, v8f c[4]) {
  const int lane = threadIdx.x & 31;
  const __bf16* arow = ldsA + (size_t)(lane & 15) * ldA;
  const __bf16* wrow = ldsWt + (size_t)(lane & 15) * KDIM;
#pragma unroll
  for (int kk = 0; kk < KDIM; kk += 32) {
    v16bf a = load_fragA(arow + kk, lane);
#pragma unroll
    for (int nt = 0; nt < 4; ++nt) {
      v16bf b = load_fragB(wrow + (size_t)nt * 16 * KDIM + kk, lane);
      c[nt] = __builtin_amdgcn_wmma_f32_16x16x32_bf16(
          false, a, false, b, (short)0, c[nt], false, false);
    }
  }
}

__device__ __forceinline__ void zero_acc(v8f c[4]) {
#pragma unroll
  for (int t = 0; t < 4; ++t)
#pragma unroll
    for (int r = 0; r < 8; ++r) c[t][r] = 0.f;
}

#define WAIT_LDS() asm volatile("s_wait_dscnt 0x0" ::: "memory")

// ---------------------------------------------------------------------------
// Utility kernels
// ---------------------------------------------------------------------------
__global__ void fill_zero_kernel(float* __restrict__ p, int n) {
  int i = blockIdx.x * blockDim.x + threadIdx.x;
  if (i < n) p[i] = 0.f;
}

__global__ void fill_zero_strided_kernel(float* __restrict__ p, int ld, int rows) {
  int i = blockIdx.x * blockDim.x + threadIdx.x;
  if (i < rows * kH) p[(size_t)(i >> 6) * ld + (i & 63)] = 0.f;
}

__global__ void degree_kernel(const int* __restrict__ tgt, float* __restrict__ cnt, int n) {
  int i = blockIdx.x * blockDim.x + threadIdx.x;
  if (i < n) atomicAdd(&cnt[tgt[i]], 1.f);
}

// ---------------------------------------------------------------------------
// Encoder: out = relu(x @ W1 + b1) @ W2 + b2   (x: [rows,2])
// 2 waves / block, 16 rows / wave. rows % 16 == 0.
// ---------------------------------------------------------------------------
__global__ void __launch_bounds__(64) encoder_kernel(
    const float* __restrict__ x, const float* __restrict__ W1,
    const float* __restrict__ b1, const float* __restrict__ W2,
    const float* __restrict__ b2, float* __restrict__ out, int ld_out, int nrows) {
  __shared__ __align__(16) __bf16 sWt[64 * 64];   // W2 transposed [n][k]
  __shared__ float sW1[128], sB1[64], sB2[64];
  __shared__ __align__(16) __bf16 sA[2][16 * 64];
  for (int i = threadIdx.x; i < 64 * 64; i += 64) {
    int k = i >> 6, n = i & 63;
    sWt[n * 64 + k] = f2bf(W2[i]);
  }
  for (int i = threadIdx.x; i < 64; i += 64) {
    sW1[i] = W1[i]; sW1[64 + i] = W1[64 + i];
    sB1[i] = b1[i]; sB2[i] = b2[i];
  }
  __syncthreads();
  const int wave = threadIdx.x >> 5, lane = threadIdx.x & 31;
  const int row0 = (blockIdx.x * 2 + wave) * 16;
  if (row0 >= nrows) return;  // wave-uniform; no syncthreads after this
  for (int i = lane; i < 16 * 64; i += 32) {
    int m = i >> 6, j = i & 63;
    const float* xr = x + (size_t)(row0 + m) * 2;
    float v = xr[0] * sW1[j] + xr[1] * sW1[64 + j] + sB1[j];
    sA[wave][i] = f2bf(fmaxf(v, 0.f));
  }
  WAIT_LDS();
  v8f c[4]; zero_acc(c);
  wmma_tile_gemm<64>(&sA[wave][0], 64, sWt, c);
  const int mofs = (lane >> 4) << 3, ncol = lane & 15;
#pragma unroll
  for (int t = 0; t < 4; ++t) {
    int col = t * 16 + ncol;
#pragma unroll
    for (int r = 0; r < 8; ++r)
      out[(size_t)(row0 + mofs + r) * ld_out + col] = c[t][r] + sB2[col];
  }
}

// ---------------------------------------------------------------------------
// Edge MLP: h = relu(relu(a*W1 + b1) @ W2 + b2), a: [E,1]; fused BN stats.
// 2 waves / block; E % 32 == 0 so all waves active (safe __syncthreads).
// ---------------------------------------------------------------------------
__global__ void __launch_bounds__(64) edge_mlp_kernel(
    const float* __restrict__ eattr, const float* __restrict__ W1,
    const float* __restrict__ b1, const float* __restrict__ W2,
    const float* __restrict__ b2, float* __restrict__ hout,
    float* __restrict__ stats) {
  __shared__ __align__(16) __bf16 sWt[64 * 64];
  __shared__ float sW1[64], sB1[64], sB2[64], sSum[64], sSsq[64];
  __shared__ __align__(16) __bf16 sA[2][16 * 64];
  for (int i = threadIdx.x; i < 64 * 64; i += 64) {
    int k = i >> 6, n = i & 63;
    sWt[n * 64 + k] = f2bf(W2[i]);
  }
  for (int i = threadIdx.x; i < 64; i += 64) {
    sW1[i] = W1[i]; sB1[i] = b1[i]; sB2[i] = b2[i];
    sSum[i] = 0.f; sSsq[i] = 0.f;
  }
  __syncthreads();
  const int wave = threadIdx.x >> 5, lane = threadIdx.x & 31;
  const int row0 = (blockIdx.x * 2 + wave) * 16;
  for (int i = lane; i < 16 * 64; i += 32) {
    int m = i >> 6, j = i & 63;
    float a = eattr[row0 + m];
    sA[wave][i] = f2bf(fmaxf(a * sW1[j] + sB1[j], 0.f));
  }
  WAIT_LDS();
  v8f c[4]; zero_acc(c);
  wmma_tile_gemm<64>(&sA[wave][0], 64, sWt, c);
  const int mofs = (lane >> 4) << 3, ncol = lane & 15;
#pragma unroll
  for (int t = 0; t < 4; ++t) {
    int col = t * 16 + ncol;
#pragma unroll
    for (int r = 0; r < 8; ++r) {
      float v = fmaxf(c[t][r] + sB2[col], 0.f);
      hout[(size_t)(row0 + mofs + r) * 64 + col] = v;
      atomicAdd(&sSum[col], v);
      atomicAdd(&sSsq[col], v * v);
    }
  }
  __syncthreads();
  if (threadIdx.x < 64) {
    atomicAdd(&stats[threadIdx.x], sSum[threadIdx.x]);
    atomicAdd(&stats[64 + threadIdx.x], sSsq[threadIdx.x]);
  }
}

// ---------------------------------------------------------------------------
// BN finalize: scale = g*rsqrt(var+eps), shift = beta - mean*scale. 1x64.
// ---------------------------------------------------------------------------
__global__ void bn_finalize_kernel(const float* __restrict__ stats,
                                   const float* __restrict__ g,
                                   const float* __restrict__ b, float invN,
                                   float* __restrict__ ss) {
  int j = threadIdx.x;
  float mean = stats[j] * invN;
  float var = fmaxf(stats[64 + j] * invN - mean * mean, 0.f);
  float sc = g[j] * rsqrtf(var + kEPS);
  ss[j] = sc;
  ss[64 + j] = b[j] - mean * sc;
}

// BN apply + convert to bf16 (feeds next GEMM's A staging directly)
__global__ void bn_apply_kernel(const float* __restrict__ h,
                                const float* __restrict__ ss,
                                __bf16* __restrict__ e, int n) {
  int i = blockIdx.x * blockDim.x + threadIdx.x;
  if (i < n) {
    int j = i & 63;
    e[i] = f2bf(h[i] * ss[j] + ss[64 + j]);
  }
}

// ---------------------------------------------------------------------------
// Message MLP: m_in = [tgt_x[etgt] | src_x[esrc] | e]  ([E,192] bf16, gathered
// into LDS), h = relu(relu(m_in @ W1 + b1) @ W2 + b2); fused BN stats.
// ---------------------------------------------------------------------------
__global__ void __launch_bounds__(64) msg_mlp_kernel(
    const int* __restrict__ esrc, const int* __restrict__ etgt,
    const float* __restrict__ tgtx, int ld_tgt,
    const float* __restrict__ srcx, int ld_src,
    const __bf16* __restrict__ ebf,
    const float* __restrict__ W1, const float* __restrict__ b1,
    const float* __restrict__ W2, const float* __restrict__ b2,
    float* __restrict__ hout, float* __restrict__ stats) {
  __shared__ __align__(16) __bf16 sWt1[64 * 192];   // 24KB
  __shared__ __align__(16) __bf16 sWt2[64 * 64];    // 8KB
  __shared__ float sB1[64], sB2[64], sSum[64], sSsq[64];
  __shared__ __align__(16) __bf16 sA[2][16 * 192];  // 12KB
  __shared__ __align__(16) __bf16 sH[2][16 * 64];   // 4KB
  for (int i = threadIdx.x; i < 192 * 64; i += 64) {
    int k = i >> 6, n = i & 63;
    sWt1[n * 192 + k] = f2bf(W1[i]);
  }
  for (int i = threadIdx.x; i < 64 * 64; i += 64) {
    int k = i >> 6, n = i & 63;
    sWt2[n * 64 + k] = f2bf(W2[i]);
  }
  for (int i = threadIdx.x; i < 64; i += 64) {
    sB1[i] = b1[i]; sB2[i] = b2[i]; sSum[i] = 0.f; sSsq[i] = 0.f;
  }
  __syncthreads();
  const int wave = threadIdx.x >> 5, lane = threadIdx.x & 31;
  const int row0 = (blockIdx.x * 2 + wave) * 16;
  // Gather + f32->bf16 convert into LDS A tile [16,192]
  for (int i = lane; i < 16 * 16; i += 32) {
    int m = i >> 4, g = i & 15, e = row0 + m;
    {
      int t = etgt[e];
      const float4 v = *reinterpret_cast<const float4*>(
          tgtx + (size_t)t * ld_tgt + g * 4);
      __bf16* d = &sA[wave][m * 192 + g * 4];
      d[0] = f2bf(v.x); d[1] = f2bf(v.y); d[2] = f2bf(v.z); d[3] = f2bf(v.w);
    }
    {
      int s = esrc[e];
      const float4 v = *reinterpret_cast<const float4*>(
          srcx + (size_t)s * ld_src + g * 4);
      __bf16* d = &sA[wave][m * 192 + 64 + g * 4];
      d[0] = f2bf(v.x); d[1] = f2bf(v.y); d[2] = f2bf(v.z); d[3] = f2bf(v.w);
    }
    {  // already bf16: 4 elems = 8 bytes
      *reinterpret_cast<unsigned long long*>(&sA[wave][m * 192 + 128 + g * 4]) =
          *reinterpret_cast<const unsigned long long*>(ebf + (size_t)e * 64 + g * 4);
    }
  }
  WAIT_LDS();
  v8f c[4]; zero_acc(c);
  wmma_tile_gemm<192>(&sA[wave][0], 192, sWt1, c);
  const int mofs = (lane >> 4) << 3, ncol = lane & 15;
#pragma unroll
  for (int t = 0; t < 4; ++t) {
    int col = t * 16 + ncol;
#pragma unroll
    for (int r = 0; r < 8; ++r)
      sH[wave][(mofs + r) * 64 + col] = f2bf(fmaxf(c[t][r] + sB1[col], 0.f));
  }
  WAIT_LDS();
  zero_acc(c);
  wmma_tile_gemm<64>(&sH[wave][0], 64, sWt2, c);
#pragma unroll
  for (int t = 0; t < 4; ++t) {
    int col = t * 16 + ncol;
#pragma unroll
    for (int r = 0; r < 8; ++r) {
      float v = fmaxf(c[t][r] + sB2[col], 0.f);
      hout[(size_t)(row0 + mofs + r) * 64 + col] = v;
      atomicAdd(&sSum[col], v);
      atomicAdd(&sSsq[col], v * v);
    }
  }
  __syncthreads();
  if (threadIdx.x < 64) {
    atomicAdd(&stats[threadIdx.x], sSum[threadIdx.x]);
    atomicAdd(&stats[64 + threadIdx.x], sSsq[threadIdx.x]);
  }
}

// Mean-scatter: acc[tgt] += BN(h)
__global__ void bn_scatter_kernel(const float* __restrict__ h,
                                  const float* __restrict__ ss,
                                  const int* __restrict__ etgt,
                                  float* __restrict__ acc, int ld, int n) {
  int i = blockIdx.x * blockDim.x + threadIdx.x;
  if (i < n) {
    int e = i >> 6, j = i & 63;
    atomicAdd(&acc[(size_t)etgt[e] * ld + j], h[i] * ss[j] + ss[64 + j]);
  }
}

__global__ void mean_relu_kernel(float* __restrict__ acc, int ld,
                                 const float* __restrict__ cnt, int rows) {
  int i = blockIdx.x * blockDim.x + threadIdx.x;
  if (i < rows * kH) {
    int r = i >> 6, j = i & 63;
    float* q = &acc[(size_t)r * ld + j];
    *q = fmaxf(*q / fmaxf(cnt[r], 1.f), 0.f);
  }
}

// ---------------------------------------------------------------------------
// Head: x[NV,256] -> relu(lin1) -> relu(lin2) -> relu(lin3) -> lin4 ->
// log_softmax. 1 wave / block (LDS: 32+8+8+8+2 KB < 64KB). NV % 16 == 0.
// ---------------------------------------------------------------------------
__global__ void __launch_bounds__(32) head_kernel(
    const float* __restrict__ xcat,
    const float* __restrict__ W1, const float* __restrict__ b1,
    const float* __restrict__ W2, const float* __restrict__ b2,
    const float* __restrict__ W3, const float* __restrict__ b3,
    const float* __restrict__ W4, const float* __restrict__ b4,
    float* __restrict__ out, int nrows) {
  __shared__ __align__(16) __bf16 sWt1[64 * 256];
  __shared__ __align__(16) __bf16 sWt2[64 * 64];
  __shared__ __align__(16) __bf16 sWt3[64 * 64];
  __shared__ float sB1[64], sB2[64], sB3[64];
  __shared__ __align__(16) __bf16 sA[16 * 256];
  __shared__ __align__(16) __bf16 sH[16 * 64];
  for (int i = threadIdx.x; i < 256 * 64; i += 32) {
    int k = i >> 6, n = i & 63;
    sWt1[n * 256 + k] = f2bf(W1[i]);
  }
  for (int i = threadIdx.x; i < 64 * 64; i += 32) {
    int k = i >> 6, n = i & 63;
    sWt2[n * 64 + k] = f2bf(W2[i]);
    sWt3[n * 64 + k] = f2bf(W3[i]);
  }
  for (int i = threadIdx.x; i < 64; i += 32) {
    sB1[i] = b1[i]; sB2[i] = b2[i]; sB3[i] = b3[i];
  }
  __syncthreads();
  const int lane = threadIdx.x & 31;
  const int row0 = blockIdx.x * 16;
  if (row0 >= nrows) return;
  for (int i = lane; i < 16 * 64; i += 32) {
    int m = i >> 6, g = i & 63;
    const float4 v = *reinterpret_cast<const float4*>(
        xcat + (size_t)(row0 + m) * 256 + g * 4);
    __bf16* d = &sA[m * 256 + g * 4];
    d[0] = f2bf(v.x); d[1] = f2bf(v.y); d[2] = f2bf(v.z); d[3] = f2bf(v.w);
  }
  WAIT_LDS();
  const int mofs = (lane >> 4) << 3, ncol = lane & 15;
  v8f c[4]; zero_acc(c);
  wmma_tile_gemm<256>(&sA[0], 256, sWt1, c);
#pragma unroll
  for (int t = 0; t < 4; ++t) {
    int col = t * 16 + ncol;
#pragma unroll
    for (int r = 0; r < 8; ++r)
      sH[(mofs + r) * 64 + col] = f2bf(fmaxf(c[t][r] + sB1[col], 0.f));
  }
  WAIT_LDS();
  zero_acc(c);
  wmma_tile_gemm<64>(&sH[0], 64, sWt2, c);
#pragma unroll
  for (int t = 0; t < 4; ++t) {
    int col = t * 16 + ncol;
#pragma unroll
    for (int r = 0; r < 8; ++r)
      sH[(mofs + r) * 64 + col] = f2bf(fmaxf(c[t][r] + sB2[col], 0.f));
  }
  WAIT_LDS();
  zero_acc(c);
  wmma_tile_gemm<64>(&sH[0], 64, sWt3, c);
  // h3 (f32) into the now-dead sA slab
  float* sO = reinterpret_cast<float*>(&sA[0]);
#pragma unroll
  for (int t = 0; t < 4; ++t) {
    int col = t * 16 + ncol;
#pragma unroll
    for (int r = 0; r < 8; ++r)
      sO[(mofs + r) * 64 + col] = fmaxf(c[t][r] + sB3[col], 0.f);
  }
  WAIT_LDS();
  if (lane < 16) {
    float l0 = b4[0], l1 = b4[1];
#pragma unroll 8
    for (int k = 0; k < 64; ++k) {
      float hv = sO[lane * 64 + k];
      l0 += hv * W4[k * 2 + 0];
      l1 += hv * W4[k * 2 + 1];
    }
    float mx = fmaxf(l0, l1);
    float lse = mx + logf(expf(l0 - mx) + expf(l1 - mx));
    out[(size_t)(row0 + lane) * 2 + 0] = l0 - lse;
    out[(size_t)(row0 + lane) * 2 + 1] = l1 - lse;
  }
}

// ---------------------------------------------------------------------------
// Host orchestration
// ---------------------------------------------------------------------------
// Input layout assumption: top-level setup_inputs() dict in insertion order
// (vf, cf, eiv, eav, eic, eac), then params flattened as a JAX pytree
// (dict keys sorted alphabetically, lists/tuples in order):
//   con_enc(4), lc_edge(18), lc_nn(18), lin1(2), lin2(2), lin3(2), lin4(2),
//   lv_edge(18), lv_nn(18), var_enc(4)   -> 88 leaves, 94 inputs total.
enum {
  IN_VF = 0, IN_CF = 1, IN_EIV = 2, IN_EAV = 3, IN_EIC = 4, IN_EAC = 5,
  P_CON_ENC = 6,   // W1,b1,W2,b2
  P_LC_EDGE = 10,  // 3 x (W1,b1,W2,b2,g,beta)
  P_LC_NN   = 28,  // 3 x (W1,b1,W2,b2,g,beta)
  P_LIN1 = 46, P_LIN2 = 48, P_LIN3 = 50, P_LIN4 = 52,
  P_LV_EDGE = 54, P_LV_NN = 72,
  P_VAR_ENC = 90
};

extern "C" void kernel_launch(void* const* d_in, const int* in_sizes, int n_in,
                              void* d_out, int out_size, void* d_ws, size_t ws_size,
                              hipStream_t stream) {
  (void)in_sizes; (void)n_in; (void)out_size; (void)ws_size;
  auto F = [&](int i) { return (const float*)d_in[i]; };
  const float* vf = F(IN_VF);
  const float* cf = F(IN_CF);
  const int* eiv = (const int*)d_in[IN_EIV];
  const float* eav = F(IN_EAV);
  const int* eic = (const int*)d_in[IN_EIC];
  const float* eac = F(IN_EAC);
  float* out = (float*)d_out;

  // workspace carve (~193 MB)
  char* p = (char*)d_ws;
  auto carve = [&](size_t bytes) -> char* {
    char* r = p;
    p += (bytes + 255) & ~(size_t)255;
    return r;
  };
  float*  xcat  = (float*)carve((size_t)kNV * 256 * 4);  // xv layer slices (ld 256)
  float*  xc    = (float*)carve((size_t)kNC * 64 * 4);
  float*  hbuf  = (float*)carve((size_t)kE * 64 * 4);    // pre-BN edge hidden
  __bf16* ebuf  = (__bf16*)carve((size_t)kE * 64 * 2);   // post-BN edge emb (bf16)
  float*  cnt_c = (float*)carve((size_t)kNC * 4);
  float*  cnt_v = (float*)carve((size_t)kNV * 4);
  float*  stats = (float*)carve(128 * 4);
  float*  ss    = (float*)carve(128 * 4);

  // degrees (constant across layers)
  fill_zero_kernel<<<(kNC + 255) / 256, 256, 0, stream>>>(cnt_c, kNC);
  fill_zero_kernel<<<(kNV + 255) / 256, 256, 0, stream>>>(cnt_v, kNV);
  degree_kernel<<<(kE + 255) / 256, 256, 0, stream>>>(eiv + kE, cnt_c, kE);
  degree_kernel<<<(kE + 255) / 256, 256, 0, stream>>>(eic + kE, cnt_v, kE);

  // encoders: xv0 -> xcat slice 0 (ld 256), xc (ld 64)
  encoder_kernel<<<(kNV + 31) / 32, 64, 0, stream>>>(
      vf, F(P_VAR_ENC + 0), F(P_VAR_ENC + 1), F(P_VAR_ENC + 2), F(P_VAR_ENC + 3),
      xcat, 256, kNV);
  encoder_kernel<<<(kNC + 31) / 32, 64, 0, stream>>>(
      cf, F(P_CON_ENC + 0), F(P_CON_ENC + 1), F(P_CON_ENC + 2), F(P_CON_ENC + 3),
      xc, 64, kNC);

  auto bipartite = [&](const int* eidx, const float* eattr,
                       const float* tgtx, int ldt, const float* srcx, int lds_,
                       int edge_p, int nn_p, float* outacc, int ldo,
                       int ntgt, const float* cnt) {
    const int* esrc = eidx;
    const int* etgt = eidx + kE;
    // edge embedding + BN
    fill_zero_kernel<<<1, 128, 0, stream>>>(stats, 128);
    edge_mlp_kernel<<<kE / 32, 64, 0, stream>>>(
        eattr, F(edge_p + 0), F(edge_p + 1), F(edge_p + 2), F(edge_p + 3),
        hbuf, stats);
    bn_finalize_kernel<<<1, 64, 0, stream>>>(stats, F(edge_p + 4), F(edge_p + 5),
                                             1.0f / kE, ss);
    bn_apply_kernel<<<(kE * 64) / 256, 256, 0, stream>>>(hbuf, ss, ebuf, kE * 64);
    // gather-concat-MLP + BN
    fill_zero_kernel<<<1, 128, 0, stream>>>(stats, 128);
    msg_mlp_kernel<<<kE / 32, 64, 0, stream>>>(
        esrc, etgt, tgtx, ldt, srcx, lds_, ebuf,
        F(nn_p + 0), F(nn_p + 1), F(nn_p + 2), F(nn_p + 3), hbuf, stats);
    bn_finalize_kernel<<<1, 64, 0, stream>>>(stats, F(nn_p + 4), F(nn_p + 5),
                                             1.0f / kE, ss);
    // mean scatter + relu
    fill_zero_strided_kernel<<<(ntgt * 64 + 255) / 256, 256, 0, stream>>>(
        outacc, ldo, ntgt);
    bn_scatter_kernel<<<(kE * 64) / 256, 256, 0, stream>>>(
        hbuf, ss, etgt, outacc, ldo, kE * 64);
    mean_relu_kernel<<<(ntgt * 64 + 255) / 256, 256, 0, stream>>>(
        outacc, ldo, cnt, ntgt);
  };

  for (int l = 0; l < 3; ++l) {
    // xc = relu(bipartite(xv, xc, eiv, eav, ...))
    bipartite(eiv, eav, /*tgt*/ xc, 64, /*src*/ xcat + 64 * l, 256,
              P_LV_EDGE + 6 * l, P_LV_NN + 6 * l, /*out*/ xc, 64, kNC, cnt_c);
    // xv_{l+1} = relu(bipartite(xc, xv, eic, eac, ...))
    bipartite(eic, eac, /*tgt*/ xcat + 64 * l, 256, /*src*/ xc, 64,
              P_LC_EDGE + 6 * l, P_LC_NN + 6 * l, /*out*/ xcat + 64 * (l + 1),
              256, kNV, cnt_v);
  }

  head_kernel<<<kNV / 16, 32, 0, stream>>>(
      xcat, F(P_LIN1), F(P_LIN1 + 1), F(P_LIN2), F(P_LIN2 + 1),
      F(P_LIN3), F(P_LIN3 + 1), F(P_LIN4), F(P_LIN4 + 1), out, kNV);
}
